// Conv2d_layer_36584531427330
// MI455X (gfx1250) — matrix operand order repaired and checked by reading, not compiled, not run on hardware
//
#include <hip/hip_runtime.h>
#include <stdint.h>

// ------------------------------------------------------------------
// StyleGAN2 upsample-conv block for MI455X (gfx1250, wave32, WMMA).
// Pass 1: stride-2 transposed 3x3 conv as 4 parity-plane implicit GEMMs.
//   - bf16 WMMA (v_wmma_f32_16x16x32_bf16), f32 accumulation
//   - double-buffered LDS tiles filled by GLOBAL_LOAD_ASYNC_TO_LDS_B32,
//     overlapped with compute via partial s_wait_asynccnt (<= group size)
// Pass 2: depthwise separable 4x4 FIR + bias + leaky_relu*sqrt(2).
// ------------------------------------------------------------------

#define BATCH 32
#define IC    512
#define OC    256
#define IH    32
#define IW    32
#define OHW   65     // conv-transpose output H/W
#define FHW   64     // final output H/W
#define KC    32     // input channels per WMMA K-chunk
#define A_ST  34     // LDS row stride (floats), padded vs bank conflicts
#define B_ST  34
#define NCHK  16     // IC / KC
#define GROUP 24     // async instructions issued per wave per chunk (8 A + 16 B)
#define GAIN  0.014731391274719738f   // 1/sqrt(512*9)
#define RSQ2  1.4142135623730951f

#if defined(__HIP_DEVICE_COMPILE__) && __has_builtin(__builtin_amdgcn_global_load_async_to_lds_b32)
#define HAVE_ASYNC 1
#else
#define HAVE_ASYNC 0
#endif

#if HAVE_ASYNC
#if __has_builtin(__builtin_amdgcn_s_wait_asynccnt)
#define WAIT_ASYNC(n) __builtin_amdgcn_s_wait_asynccnt(n)
#else
#define WAIT_ASYNC(n) asm volatile("s_wait_asynccnt %0" ::"i"(n) : "memory")
#endif
#else
#define WAIT_ASYNC(n) ((void)0)
#endif

typedef __attribute__((ext_vector_type(16))) __bf16 v16bf;
typedef __attribute__((ext_vector_type(8)))  float  v8f;

union AOp { v16bf v; uint32_t u[8]; };
union Acc { v8f  v; float    f[8]; };

// round-to-nearest-even f32 -> bf16, packed pair (lo in [15:0], hi in [31:16])
static __device__ __forceinline__ uint32_t pk_bf16(float lo, float hi) {
  uint32_t a = __float_as_uint(lo);
  uint32_t b = __float_as_uint(hi);
  a += 0x7fffu + ((a >> 16) & 1u);
  b += 0x7fffu + ((b >> 16) & 1u);
  return (a >> 16) | (b & 0xffff0000u);
}

static __device__ __forceinline__ int iclamp(int v, int lo, int hi) {
  return v < lo ? lo : (v > hi ? hi : v);
}

#if HAVE_ASYNC
typedef __attribute__((address_space(1))) int gint_t;  // global (64-bit ptr)
typedef __attribute__((address_space(3))) int lint_t;  // LDS (32-bit ptr)

static __device__ __forceinline__ void async_g2l_b32(const float* g, const float* l) {
  // signature (probe-confirmed): (global int*, lds int*, imm offset, imm cpol)
  __builtin_amdgcn_global_load_async_to_lds_b32(
      (gint_t*)(uintptr_t)g, (lint_t*)(unsigned)(uintptr_t)l, 0, 0);
}
#endif

// ------------------------------------------------------------------
// Kernel 1: upsampling conv as implicit GEMM per output-parity plane.
// Block tile: 64 oc x 128 px (8 rows x 16 cols of one parity plane).
// 8 waves; each wave owns a 32x32 tile = 2x2 WMMA accumulators.
// Software pipeline: stage(c+1) async -> wait(<=GROUP) -> barrier ->
// compute(c); double-buffered LDS.
// ------------------------------------------------------------------
__global__ __launch_bounds__(256) void upconv_wmma(
    const float* __restrict__ x,    // (32, 512, 32, 32)
    const float* __restrict__ w,    // (256, 512, 3, 3)
    float* __restrict__ y)          // (32, 256, 65, 65) workspace
{
  __shared__ __align__(16) float Alds[2][64 * A_ST];    // 64 oc  x 32 k
  __shared__ __align__(16) float Blds[2][128 * B_ST];   // 128 px x 32 k

  const int tid   = threadIdx.x;
  const int b     = blockIdx.z;
  const int plane = blockIdx.y >> 2;      // 0..3 : (py,px) parity plane
  const int ocb   = blockIdx.y & 3;       // 64-channel tile index
  const int py = plane >> 1, px = plane & 1;
  const int ph = 33 - py, pw = 33 - px;   // parity-plane dims
  const int tr = blockIdx.x / 3, tc = blockIdx.x % 3;
  const int r0 = tr * 8, c0 = tc * 16;
  if (r0 >= ph || c0 >= pw) return;       // uniform early-out

  const int ntX = 2 - px;                 // taps in x for this parity
  const int ntY = 2 - py;
  const int total = ntY * ntX * NCHK;     // flattened (tY,tX,icc) chunks

  const int lane = tid & 31;
  const int wv   = tid >> 5;              // 8 wave32 waves
  const int woc  = wv & 1;                // oc rows  [woc*32, +32)
  const int wpx  = wv >> 1;               // px cols  [wpx*32, +32)
  const int half = lane >> 4;
  const int l15  = lane & 15;

  // staging assignments: A: 64x32 elems -> 8/thread; B: 128x32 -> 16/thread
  const int a_oc = tid >> 2;
  const int a_k0 = (tid & 3) * 8;
  const int b_px = tid >> 1;
  const int b_k0 = (tid & 1) * 16;
  const int srow = r0 + (b_px >> 4);      // staging pixel plane coords
  const int scol = c0 + (b_px & 15);

  // this lane's two B pixel columns (for validity masks in compute)
  const int n0  = wpx * 32 + l15;
  const int rr0 = r0 + (n0 >> 4),        cc0 = c0 + (n0 & 15);
  const int rr1 = r0 + ((n0 + 16) >> 4), cc1 = c0 + ((n0 + 16) & 15);

  Acc acc[2][2];
  acc[0][0].v = {}; acc[0][1].v = {}; acc[1][0].v = {}; acc[1][1].v = {};

  // chunk c -> tap (tY,tX) + ic offset; dy/dx: input offset, wky/wkx: flipped tap
  auto tap_of = [&](int c, int& dy, int& dx, int& wky, int& wkx, int& icc) {
    const int tY = c / (ntX * NCHK);
    const int rm = c % (ntX * NCHK);
    const int tX = rm / NCHK;
    icc = (rm % NCHK) * KC;
    dy  = py ? 0 : (tY - 1);  wky = py ? 1 : (2 - 2 * tY);
    dx  = px ? 0 : (tX - 1);  wkx = px ? 1 : (2 - 2 * tX);
  };

  auto stage = [&](int c, int bsel) {
    int dy, dx, wky, wkx, icc;
    tap_of(c, dy, dx, wky, wkx, icc);
    // ---- weight tile A[oc][k]: W[ocg, icc+k, wky, wkx] (stride 9 in ic) ----
    const float* ga = w + ((size_t)(ocb * 64 + a_oc) * IC + icc + a_k0) * 9 + wky * 3 + wkx;
    float* la = &Alds[bsel][a_oc * A_ST + a_k0];
    // ---- input tile B[pix][k]: X[b, icc+k, i, j], address clamped in-bounds;
    //      invalid pixels are zero-masked at operand build ----
    const int i = iclamp(srow + dy, 0, IH - 1);
    const int j = iclamp(scol + dx, 0, IW - 1);
    const float* gb = x + (((size_t)b * IC + icc + b_k0) * IH + i) * IW + j;
    float* lb = &Blds[bsel][b_px * B_ST + b_k0];
#if HAVE_ASYNC
#pragma unroll
    for (int e = 0; e < 8; ++e)  async_g2l_b32(ga + e * 9, la + e);
#pragma unroll
    for (int e = 0; e < 16; ++e) async_g2l_b32(gb + e * (IH * IW), lb + e);
#else
#pragma unroll
    for (int e = 0; e < 8; ++e)  la[e] = ga[(size_t)e * 9];
#pragma unroll
    for (int e = 0; e < 16; ++e) lb[e] = gb[(size_t)e * (IH * IW)];
#endif
  };

  auto compute = [&](int c, int bsel) {
    int dy, dx, wky, wkx, icc;
    tap_of(c, dy, dx, wky, wkx, icc);
    // per-lane validity masks for the two pixel columns read by this lane
    const int i0 = rr0 + dy, j0 = cc0 + dx, i1 = rr1 + dy, j1 = cc1 + dx;
    const uint32_t m0 = (rr0 < ph && cc0 < pw && i0 >= 0 && i0 < IH && j0 >= 0 && j0 < IW)
                            ? 0xffffffffu : 0u;
    const uint32_t m1 = (rr1 < ph && cc1 < pw && i1 >= 0 && i1 < IH && j1 >= 0 && j1 < IW)
                            ? 0xffffffffu : 0u;
    AOp Aop[2], Bop[2];
#pragma unroll
    for (int v = 0; v < 8; ++v) {
      // A 16x32: lanes0-15 v0-3 K=2v,+1 / v4-7 K=16+2(v-4); lanes16-31 K+=8
      const int ka = (v < 4) ? (2 * v + 8 * half) : (16 + 2 * (v - 4) + 8 * half);
      // B 32x16: same K mapping as A
      const int kb = (v < 4) ? (2 * v + 8 * half) : (16 + 2 * (v - 4) + 8 * half);
      const int m0r = woc * 32 + l15;
      const float2 fa0 = *(const float2*)&Alds[bsel][m0r * A_ST + ka];
      const float2 fa1 = *(const float2*)&Alds[bsel][(m0r + 16) * A_ST + ka];
      const float2 fb0 = *(const float2*)&Blds[bsel][n0 * B_ST + kb];
      const float2 fb1 = *(const float2*)&Blds[bsel][(n0 + 16) * B_ST + kb];
      Aop[0].u[v] = pk_bf16(fa0.x, fa0.y);
      Aop[1].u[v] = pk_bf16(fa1.x, fa1.y);
      Bop[0].u[v] = pk_bf16(fb0.x, fb0.y) & m0;
      Bop[1].u[v] = pk_bf16(fb1.x, fb1.y) & m1;
    }
#pragma unroll
    for (int mb = 0; mb < 2; ++mb)
#pragma unroll
      for (int nb = 0; nb < 2; ++nb)
        acc[mb][nb].v = __builtin_amdgcn_wmma_f32_16x16x32_bf16(
            false, Aop[mb].v, false, Bop[nb].v,
            (short)0, acc[mb][nb].v, false, false);
  };

  // ---- software pipeline: prefetch chunk c+1 while computing chunk c ----
  stage(0, 0);
  for (int c = 0; c < total; ++c) {
    __syncthreads();                 // buffer (c+1)&1 free: compute(c-1) done
    if (c + 1 < total) {
      stage(c + 1, (c + 1) & 1);
      WAIT_ASYNC(GROUP);             // only group c must have landed
    } else {
      WAIT_ASYNC(0);
    }
    __syncthreads();                 // group c visible to all waves
    compute(c, c & 1);
  }

  // ---- epilogue: scale by gain, scatter to Y (parity plane -> 65x65) ----
#pragma unroll
  for (int mb = 0; mb < 2; ++mb)
#pragma unroll
    for (int nb = 0; nb < 2; ++nb)
#pragma unroll
      for (int r = 0; r < 8; ++r) {
        const int m  = woc * 32 + mb * 16 + r + 8 * half;  // C/D layout
        const int n  = wpx * 32 + nb * 16 + l15;
        const int rr = r0 + (n >> 4), cc = c0 + (n & 15);
        if (rr < ph && cc < pw) {
          const int oh = 2 * rr + py, ow = 2 * cc + px;
          y[(((size_t)b * OC + ocb * 64 + m) * OHW + oh) * OHW + ow] =
              acc[mb][nb].f[r] * GAIN;
        }
      }
}

// ------------------------------------------------------------------
// Kernel 2: depthwise separable 4x4 FIR (pad 1, gain up^2 folded in),
// + bias + leaky_relu(0.2) * sqrt(2).  F = [1,3,3,1]/8 * 2 per axis.
// ------------------------------------------------------------------
__global__ __launch_bounds__(256) void fir_bias_lrelu(
    const float* __restrict__ yin,   // (32, 256, 65, 65)
    const float* __restrict__ bias,  // (256,)
    float* __restrict__ out)         // (32, 256, 64, 64)
{
  __shared__ float t[19][20];
  const int tid = threadIdx.x;
  const int b = blockIdx.z, oc = blockIdx.y;
  const int ty0 = (blockIdx.x >> 2) * 16, tx0 = (blockIdx.x & 3) * 16;
  const float* src = yin + ((size_t)b * OC + oc) * OHW * OHW;

  for (int idx = tid; idx < 19 * 19; idx += 256) {
    const int ly = idx / 19, lx = idx % 19;
    const int sy = ty0 - 1 + ly, sx = tx0 - 1 + lx;
    float v = 0.0f;
    if (sy >= 0 && sy < OHW && sx >= 0 && sx < OHW) v = src[sy * OHW + sx];
    t[ly][lx] = v;
  }
  __syncthreads();

  const int ly = tid >> 4, lx = tid & 15;
  const float F0 = 0.25f, F1 = 0.75f;   // [1,3,3,1]/8 * up: F*F sums to up^2
  float s = 0.0f;
#pragma unroll
  for (int a = 0; a < 4; ++a) {
    const float rs = t[ly + a][lx] * F0 + t[ly + a][lx + 1] * F1 +
                     t[ly + a][lx + 2] * F1 + t[ly + a][lx + 3] * F0;
    s += rs * ((a == 0 || a == 3) ? F0 : F1);
  }
  s += bias[oc];
  s = (s >= 0.0f ? s : 0.2f * s) * RSQ2;
  out[(((size_t)b * OC + oc) * FHW + (ty0 + ly)) * FHW + (tx0 + lx)] = s;
}

// ------------------------------------------------------------------
extern "C" void kernel_launch(void* const* d_in, const int* in_sizes, int n_in,
                              void* d_out, int out_size, void* d_ws, size_t ws_size,
                              hipStream_t stream) {
  (void)in_sizes; (void)n_in; (void)out_size; (void)ws_size;
  const float* x    = (const float*)d_in[0];
  const float* w    = (const float*)d_in[1];
  const float* bias = (const float*)d_in[2];
  float* out = (float*)d_out;
  float* yws = (float*)d_ws;   // needs 32*256*65*65 floats (~138 MB)

  // 15 pixel tiles (5 row x 3 col over max 33x33 plane) x (4 planes * 4 oc tiles) x batch
  dim3 g1(15, 16, BATCH), b1(256);
  hipLaunchKernelGGL(upconv_wmma, g1, b1, 0, stream, x, w, yws);

  dim3 g2(16, OC, BATCH), b2(256);
  hipLaunchKernelGGL(fir_bias_lrelu, g2, b2, 0, stream, yws, bias, out);
}
